// SelfAttention_34660386079354
// MI455X (gfx1250) — matrix operand order repaired and checked by reading, not compiled, not run on hardware
//
#include <hip/hip_runtime.h>
#include <hip/hip_bf16.h>

// ---------------------------------------------------------------------------
// Self-attention for MI455X (gfx1250), wave32 + v_wmma_f32_16x16x32_bf16.
// Shapes: B=8, CI=512, CO=256, N=T*H*W=2048.
// bf16 operands / f32 accumulation on the WMMA pipe; flash-style online
// softmax (NxN scores never hit global memory); b128 operand loads; LDS
// staged + wave-shared activation tiles for the projection GEMMs.
// ---------------------------------------------------------------------------

typedef __bf16 bf16;
typedef __attribute__((ext_vector_type(16))) __bf16 v16bf;
typedef __attribute__((ext_vector_type(8)))  __bf16 bf16x8;
typedef __attribute__((ext_vector_type(4)))  __bf16 bf16x4;
typedef __attribute__((ext_vector_type(2)))  __bf16 bf16x2;
typedef __attribute__((ext_vector_type(8)))  float  v8f;
typedef __attribute__((ext_vector_type(4)))  float  f32x4;

#define WMMA_BF16(A, Bm, C) \
  __builtin_amdgcn_wmma_f32_16x16x32_bf16(false, (A), false, (Bm), (short)0, (C), false, false)

constexpr int Bb = 8, CI = 512, CO = 256, NN = 2048;
constexpr float SM_SCALE = 0.0625f; // 1/sqrt(CO)

// Fragment = two contiguous 16-byte (8 x bf16) loads, in this code's 16-bit
// operand order, same for A and B: K = 16h+0..15.
__device__ __forceinline__ v16bf load_frag2(const bf16* p0, const bf16* p1) {
  bf16x8 lo = *(const bf16x8*)p0;
  bf16x8 hi = *(const bf16x8*)p1;
  return __builtin_shufflevector(lo, hi, 0, 1, 2, 3, 4, 5, 6, 7,
                                         8, 9, 10, 11, 12, 13, 14, 15);
}

// A-fragment from a row of fp32 weights (K-contiguous): 4x b128 loads + cvt.
__device__ __forceinline__ v16bf a_frag_f32(const float* row, int half) {
  f32x4 w0 = *(const f32x4*)(row + 16 * half);
  f32x4 w1 = *(const f32x4*)(row + 16 * half + 4);
  f32x4 w2 = *(const f32x4*)(row + 16 * half + 8);
  f32x4 w3 = *(const f32x4*)(row + 16 * half + 12);
  v16bf a;
  #pragma unroll
  for (int e = 0; e < 4; ++e) {
    a[e]      = (bf16)w0[e];
    a[4 + e]  = (bf16)w1[e];
    a[8 + e]  = (bf16)w2[e];
    a[12 + e] = (bf16)w3[e];
  }
  return a;
}

// ---------------------------------------------------------------------------
// Kernel 1: K/Q/V projections.  4 waves/block share one staged x-tile via LDS;
// wave w computes co-tile (strip*64 + w*16).  D[16co x 16n] += W x X.
// K,Q stored transposed as [n][c] bf16 (one b128 store per lane);
// V stored as [c][m] bf16.
// ---------------------------------------------------------------------------
__global__ __launch_bounds__(128) void
kqv_kernel(const float* __restrict__ x, const float* __restrict__ query,
           const float* __restrict__ kw, const float* __restrict__ kb,
           const float* __restrict__ vw, const float* __restrict__ vb,
           const float* __restrict__ qw, const float* __restrict__ qb,
           bf16* __restrict__ Kt, bf16* __restrict__ Qt, bf16* __restrict__ Vc) {
  __shared__ bf16 xt[16][36];          // [n][ci] transposed tile, 72B rows

  const int tid  = threadIdx.x;
  const int wave = tid >> 5;
  const int lane = tid & 31;
  const int half = lane >> 4;
  const int l16  = lane & 15;
  const int n0   = blockIdx.x * 16;
  const int co0  = blockIdx.y * 64 + wave * 16;
  const int b    = blockIdx.z / 3;
  const int mode = blockIdx.z % 3;     // 0=K, 1=Q, 2=V

  const float* w;  const float* bias; const float* src;
  if (mode == 0)      { w = kw; bias = kb; src = x; }
  else if (mode == 1) { w = qw; bias = qb; src = query; }
  else                { w = vw; bias = vb; src = x; }

  const float* xb = src + (size_t)b * CI * NN;

  const int scol = tid & 15;           // staging: n column
  const int srow = (tid >> 4) * 4;     // staging: 4 ci rows per thread

  v8f acc = {};
  #pragma unroll 1
  for (int c0 = 0; c0 < CI; c0 += 32) {
    // global loads first: overlap with previous iteration's LDS reads + WMMA
    v16bf a = a_frag_f32(w + (size_t)(co0 + l16) * CI + c0, half);
    const float* sp = xb + (size_t)(c0 + srow) * NN + n0 + scol;
    const float f0 = sp[0];
    const float f1 = sp[NN];
    const float f2 = sp[2 * NN];
    const float f3 = sp[3 * NN];

    __syncthreads();                   // prior reads of xt complete
    {  // 32ci x 16n fp32 -> bf16, transposed: one aligned ds_store_b64
      bf16x4 pk;
      pk[0] = (bf16)f0; pk[1] = (bf16)f1; pk[2] = (bf16)f2; pk[3] = (bf16)f3;
      *(bf16x4*)&xt[scol][srow] = pk;
    }
    __syncthreads();

    v16bf bfr;
    #pragma unroll
    for (int j = 0; j < 8; ++j) {
      bf16x2 p = *(const bf16x2*)&xt[l16][2 * j + 16 * half];
      bfr[2 * j] = p.x; bfr[2 * j + 1] = p.y;
    }
    acc = WMMA_BF16(a, bfr, acc);
  }

  // ---- epilogue: a lane's 8 values are 8 consecutive co = co0+8h+0..7 ----
  const f32x4 bias_lo = *(const f32x4*)(bias + co0 + 8 * half);
  const f32x4 bias_hi = *(const f32x4*)(bias + co0 + 8 * half + 4);
  bf16x8 pk;
  #pragma unroll
  for (int e = 0; e < 4; ++e) {
    pk[e]     = (bf16)(acc[e] + bias_lo[e]);
    pk[4 + e] = (bf16)(acc[4 + e] + bias_hi[e]);
  }
  if (mode == 2) {                     // V: [c][m] rows, per-row b16 stores
    bf16* base = Vc + ((size_t)b * CO + co0 + 8 * half) * NN + n0 + l16;
    #pragma unroll
    for (int r = 0; r < 8; ++r) base[(size_t)r * NN] = pk[r];
  } else {                             // K/Q: [n][c] -> single b128 store
    bf16* dst = (mode == 0) ? Kt : Qt;
    *(bf16x8*)(dst + ((size_t)b * NN + n0 + l16) * CO + co0 + 8 * half) = pk;
  }
}

// ---------------------------------------------------------------------------
// Kernel 2: flash attention.  One wave per (batch, 16-row n-tile, 128-col
// c-chunk).  K^T A-fragments preloaded (m-invariant); all global operand
// loads are b128; P relayout through padded LDS (conflict-free b128 reads);
// next-step Q rows prefetched (global_prefetch_b8).
// ---------------------------------------------------------------------------
__global__ __launch_bounds__(32) void
attn_kernel(const bf16* __restrict__ Kt, const bf16* __restrict__ Qt,
            const bf16* __restrict__ Vc, float* __restrict__ out3) {
  __shared__ bf16 pbuf[16][40];        // 80B rows: b128 readback conflict-free

  const int lane  = threadIdx.x & 31;
  const int half  = lane >> 4;
  const int l16   = lane & 15;
  const int n0    = blockIdx.x * 16;
  const int cbase = blockIdx.y * 128;
  const int b     = blockIdx.z;

  const bf16* Ktb = Kt + (size_t)b * NN * CO;
  const bf16* Qtb = Qt + (size_t)b * NN * CO;
  const bf16* Vcb = Vc + (size_t)b * CO * NN;

  // Preload all 8 A-fragments of K^T (rows n0..n0+15, K = c) -- m-invariant.
  const bf16* krow = Ktb + (size_t)(n0 + l16) * CO;
  v16bf Ka[8];
  #pragma unroll
  for (int c = 0; c < 8; ++c)
    Ka[c] = load_frag2(krow + c * 32 + 16 * half, krow + c * 32 + 16 * half + 8);

  v8f O[8];
  #pragma unroll
  for (int t = 0; t < 8; ++t) O[t] = {};
  float mrun[8], lrun[8];
  #pragma unroll
  for (int r = 0; r < 8; ++r) { mrun[r] = -3.0e38f; lrun[r] = 0.0f; }

  const bf16* q0row = Qtb + (size_t)l16 * CO + 16 * half;     // advance by m
  const bf16* q1row = Qtb + (size_t)(16 + l16) * CO + 16 * half;

  #pragma unroll 1
  for (int m0 = 0; m0 < NN; m0 += 32) {
    // ---- S tiles: rows n0..n0+15, cols m0..m0+31, K over c = 0..255 ----
    v8f S0 = {}, S1 = {};
    const bf16* q0 = q0row + (size_t)m0 * CO;
    const bf16* q1 = q1row + (size_t)m0 * CO;
    __builtin_prefetch(q0 + 32 * CO, 0, 0);   // next m-step Q rows
    __builtin_prefetch(q1 + 32 * CO, 0, 0);
    #pragma unroll 2
    for (int c = 0; c < 8; ++c) {
      v16bf b0 = load_frag2(q0 + c * 32, q0 + c * 32 + 8);
      v16bf b1 = load_frag2(q1 + c * 32, q1 + c * 32 + 8);
      S0 = WMMA_BF16(Ka[c], b0, S0);
      S1 = WMMA_BF16(Ka[c], b1, S1);
    }

    // ---- online softmax: reductions across the 16-lane half (cols = m) ----
    float alpha[8];
    #pragma unroll
    for (int r = 0; r < 8; ++r) {
      const float s0 = S0[r] * SM_SCALE, s1 = S1[r] * SM_SCALE;
      float mx = fmaxf(s0, s1);
      #pragma unroll
      for (int off = 1; off < 16; off <<= 1)
        mx = fmaxf(mx, __shfl_xor(mx, off, 32));
      const float mnew = fmaxf(mrun[r], mx);
      const float p0 = __expf(s0 - mnew);
      const float p1 = __expf(s1 - mnew);
      float rs = p0 + p1;
      #pragma unroll
      for (int off = 1; off < 16; off <<= 1)
        rs += __shfl_xor(rs, off, 32);
      alpha[r] = __expf(mrun[r] - mnew);
      lrun[r]  = lrun[r] * alpha[r] + rs;
      mrun[r]  = mnew;
      const int row = r + 8 * half;
      pbuf[row][l16]      = (bf16)p0;          // cols m0..m0+15
      pbuf[row][l16 + 16] = (bf16)p1;          // cols m0+16..m0+31
    }
    __syncthreads();

    // ---- P back out in A-operand layout: two b128 LDS reads per lane ----
    v16bf pa = load_frag2(&pbuf[l16][16 * half], &pbuf[l16][16 * half + 8]);
    __syncthreads();

    // ---- O[c-chunk] = O*alpha + P @ V^T ----
    #pragma unroll 2
    for (int t = 0; t < 8; ++t) {
      const bf16* vp = Vcb + (size_t)(cbase + 16 * t + l16) * NN + m0 + 16 * half;
      v16bf bv = load_frag2(vp, vp + 8);
      #pragma unroll
      for (int r = 0; r < 8; ++r) O[t][r] *= alpha[r];
      O[t] = WMMA_BF16(pa, bv, O[t]);
    }
  }

  // ---- normalize and store [B,N,Co] fp32 ----
  #pragma unroll
  for (int r = 0; r < 8; ++r) lrun[r] = 1.0f / lrun[r];
  #pragma unroll
  for (int t = 0; t < 8; ++t) {
    #pragma unroll
    for (int r = 0; r < 8; ++r) {
      const int row = n0 + r + 8 * half;
      out3[((size_t)b * NN + row) * CO + cbase + 16 * t + l16] = O[t][r] * lrun[r];
    }
  }
}

// ---------------------------------------------------------------------------
// Kernel 3: up-projection on the raw .view + bias + residual.
// out3 flat [n][c] re-read as [co][pos] (the torch .view).  Same LDS-shared
// staging as kernel 1: 4 waves/block, one staged B tile.
// ---------------------------------------------------------------------------
__global__ __launch_bounds__(128) void
up_kernel(const float* __restrict__ x, const float* __restrict__ upw,
          const float* __restrict__ upb, const float* __restrict__ scaling,
          const float* __restrict__ out3, float* __restrict__ y) {
  __shared__ bf16 ot[16][36];

  const int tid  = threadIdx.x;
  const int wave = tid >> 5;
  const int lane = tid & 31;
  const int half = lane >> 4;
  const int l16  = lane & 15;
  const int p0   = blockIdx.x * 16;                  // pos tile
  const int ci0  = blockIdx.y * 64 + wave * 16;      // out-channel tile
  const int b    = blockIdx.z;
  const float* o3b = out3 + (size_t)b * CO * NN;     // flat view [co][pos]
  const float sc = *scaling;

  const int scol = tid & 15;
  const int srow = (tid >> 4) * 4;

  v8f acc = {};
  #pragma unroll 1
  for (int c0 = 0; c0 < CO; c0 += 32) {
    v16bf a = a_frag_f32(upw + (size_t)(ci0 + l16) * CO + c0, half);
    const float* sp = o3b + (size_t)(c0 + srow) * NN + p0 + scol;
    const float f0 = sp[0];
    const float f1 = sp[NN];
    const float f2 = sp[2 * NN];
    const float f3 = sp[3 * NN];

    __syncthreads();
    {
      bf16x4 pk;
      pk[0] = (bf16)f0; pk[1] = (bf16)f1; pk[2] = (bf16)f2; pk[3] = (bf16)f3;
      *(bf16x4*)&ot[scol][srow] = pk;
    }
    __syncthreads();

    v16bf bv;
    #pragma unroll
    for (int j = 0; j < 8; ++j) {
      bf16x2 p = *(const bf16x2*)&ot[l16][2 * j + 16 * half];
      bv[2 * j] = p.x; bv[2 * j + 1] = p.y;
    }
    acc = WMMA_BF16(a, bv, acc);
  }

  #pragma unroll
  for (int r = 0; r < 8; ++r) {
    const int ci = ci0 + r + 8 * half;
    const size_t idx = ((size_t)b * CI + ci) * NN + p0 + l16;
    y[idx] = x[idx] + sc * (acc[r] + upb[ci]);
  }
}

// ---------------------------------------------------------------------------
extern "C" void kernel_launch(void* const* d_in, const int* in_sizes, int n_in,
                              void* d_out, int out_size, void* d_ws, size_t ws_size,
                              hipStream_t stream) {
  const float* x   = (const float*)d_in[0];
  const float* qry = (const float*)d_in[1];
  const float* kw  = (const float*)d_in[2];
  const float* kb  = (const float*)d_in[3];
  const float* vw  = (const float*)d_in[4];
  const float* vb  = (const float*)d_in[5];
  const float* qw  = (const float*)d_in[6];
  const float* qb  = (const float*)d_in[7];
  const float* uw  = (const float*)d_in[8];
  const float* ub  = (const float*)d_in[9];
  const float* sc  = (const float*)d_in[10];
  float* y = (float*)d_out;

  // Workspace layout (40 MB total):
  //   Kt  bf16 [B][N][CO]  @ 0      (8 MB)
  //   Qt  bf16 [B][N][CO]  @ 8  MB
  //   Vc  bf16 [B][CO][N]  @ 16 MB
  //   out3 f32 [B][N][CO]  @ 24 MB  (16 MB)
  char* ws = (char*)d_ws;
  bf16*  Kt   = (bf16*)(ws);
  bf16*  Qt   = (bf16*)(ws + 8388608);
  bf16*  Vc   = (bf16*)(ws + 16777216);
  float* out3 = (float*)(ws + 25165824);

  kqv_kernel<<<dim3(NN / 16, CO / 64, Bb * 3), dim3(128, 1, 1), 0, stream>>>(
      x, qry, kw, kb, vw, vb, qw, qb, Kt, Qt, Vc);
  attn_kernel<<<dim3(NN / 16, CO / 128, Bb), dim3(32, 1, 1), 0, stream>>>(
      Kt, Qt, Vc, out3);
  up_kernel<<<dim3(NN / 16, CI / 64, Bb), dim3(128, 1, 1), 0, stream>>>(
      x, uw, ub, sc, out3, y);
}
